// Attention_75926431859292
// MI455X (gfx1250) — matrix operand order repaired and checked by reading, not compiled, not run on hardware
//
#include <hip/hip_runtime.h>
#include <hip/hip_bf16.h>

typedef __attribute__((ext_vector_type(16))) __bf16 v16bf;
typedef __attribute__((ext_vector_type(8)))  float  v8f;

constexpr int kBT  = 64;          // B*T
constexpr int kJ   = 17;
constexpr int kN   = 17;
constexpr int kC   = 256;
constexpr int kH   = 8;
constexpr int kHC  = 32;          // C / H
constexpr int kRows = kBT * kJ * kN;     // 18496 = 289 * 64
constexpr int kQKV  = 5 * kC;            // 1280
constexpr int kCat  = 3 * kC;            // 768
constexpr float kScale = 0.17677669529663687f; // 32^-0.5

constexpr int kLdsPad = 40;              // 32 k + 8 pad (bf16 units)

// ---------------------------------------------------------------------------
// Shared GEMM core: block = 256 threads = 8 waves (4 M-subtiles x 2 N-subtiles),
// block tile 64(M) x 128(N), K stepped by 32 with LDS staging in bf16.
// Each wave: 1 A-frag + 4 B-frags -> 4 WMMAs per K step.
// ---------------------------------------------------------------------------

// ---------------------------------------------------------------------------
// Kernel 1: qkv = X @ W_qkv  (bf16 WMMA, f32 accum), fused interleaved split
//           q/k/vs/vv/vsv = qkv[..., c, which] * M[j,n,c]
// Grid: 289 (M blocks) * 10 (N blocks) = 2890 blocks.
// ---------------------------------------------------------------------------
__global__ __launch_bounds__(256)
void qkv_gemm_kernel(const float* __restrict__ X,
                     const float* __restrict__ Wqkv,
                     const float* __restrict__ Mm,
                     float* __restrict__ qo, float* __restrict__ ko,
                     float* __restrict__ vso, float* __restrict__ vvo,
                     float* __restrict__ vsvo)
{
    __shared__ __bf16 As[64][kLdsPad];    // [row][k]
    __shared__ __bf16 Bs[128][kLdsPad];   // [col][k]  (transposed panel)

    const int t    = threadIdx.x;
    const int lane = t & 31;
    const int w    = t >> 5;
    const int wm   = w & 3;               // M subtile 0..3
    const int wn   = w >> 2;              // N subtile 0..1
    const int mlo  = lane & 15;
    const bool hi  = lane >= 16;
    const int kgA  = hi ? 8 : 0;
    const int kgB  = hi ? 16 : 0;

    const int bm = blockIdx.x / 10;
    const int bn = blockIdx.x % 10;
    const int rb = bm * 64;               // block row base
    const int cb = bn * 128;              // block col base

    // cooperative-load coordinates
    const int arow = t >> 2;              // 0..63
    const int akc  = (t & 3) * 8;         // 0,8,16,24
    const int bcl  = t & 127;             // 0..127 (panel-local col)
    const int bkh  = (t >> 7) * 16;       // 0 or 16

    v8f acc[4] = {};

    for (int kt = 0; kt < kC; kt += 32) {
        __syncthreads();
        // stage A panel: 64 rows x 32 k
        {
            const float* src = X + (size_t)(rb + arow) * kC + kt + akc;
#pragma unroll
            for (int i = 0; i < 8; ++i)
                As[arow][akc + i] = (__bf16)src[i];
            const int ktn = (kt + 32 < kC) ? kt + 32 : 0;
            __builtin_prefetch(X + (size_t)(rb + arow) * kC + ktn + akc, 0, 1);
        }
        // stage B panel (transposed): 32 k x 128 cols
        {
#pragma unroll
            for (int i = 0; i < 16; ++i)
                Bs[bcl][bkh + i] =
                    (__bf16)Wqkv[(size_t)(kt + bkh + i) * kQKV + cb + bcl];
            const int ktn = (kt + 32 < kC) ? kt + 32 : 0;
            __builtin_prefetch(Wqkv + (size_t)(ktn + bkh) * kQKV + cb + bcl, 0, 1);
        }
        __syncthreads();

        // A fragment (reused by 4 WMMAs)
        v16bf a;
#pragma unroll
        for (int e = 0; e < 16; ++e) {
            const int ka = (e < 8) ? (kgA + e) : (16 + kgA + (e - 8));
            a[e] = As[wm * 16 + mlo][ka];
        }
#pragma unroll
        for (int f = 0; f < 4; ++f) {
            v16bf b;
#pragma unroll
            for (int e = 0; e < 16; ++e)
                b[e] = Bs[wn * 64 + f * 16 + mlo][(e < 8) ? (kgA + e) : (16 + kgA + (e - 8))];
            acc[f] = __builtin_amdgcn_wmma_f32_16x16x32_bf16(
                false, a, false, b, (short)0, acc[f], false, false);
        }
    }

    // epilogue: interleaved split (col = c*5 + which) and M-gate
#pragma unroll
    for (int f = 0; f < 4; ++f) {
        const int col = cb + wn * 64 + f * 16 + mlo;
        const int cc = col / 5;
        const int which = col % 5;
        float* dst = (which == 0) ? qo
                   : (which == 1) ? ko
                   : (which == 2) ? vso
                   : (which == 3) ? vvo : vsvo;
#pragma unroll
        for (int i = 0; i < 8; ++i) {
            const int row = rb + wm * 16 + i + (hi ? 8 : 0);
            const int nn  = row % kN;
            const int jj  = (row / kN) % kJ;
            const float mv = Mm[(size_t)(jj * kN + nn) * kC + cc];
            dst[(size_t)row * kC + cc] = acc[f][i] * mv;
        }
    }
}

// ---------------------------------------------------------------------------
// Kernel 2: spatial attention. One wave per (b,h,n); lane j owns row j.
// ---------------------------------------------------------------------------
__global__ __launch_bounds__(256)
void att_s_kernel(const float* __restrict__ Q, const float* __restrict__ Kt,
                  const float* __restrict__ As, const float* __restrict__ adjs,
                  float* __restrict__ att1, float* __restrict__ att2)
{
    const int lane = threadIdx.x & 31;
    const int u = blockIdx.x * 8 + (threadIdx.x >> 5);  // (b*H + h)*N + n
    const int n = u % kN;
    const int h = (u / kN) % kH;
    const int b = u / (kN * kH);
    const int j = lane;
    if (j >= kJ) return;

    const float* qrow = Q + (size_t)((b * kJ + j) * kN + n) * kC + h * kHC;
    float s[kJ];
    for (int k2 = 0; k2 < kJ; ++k2) {
        const float* krow = Kt + (size_t)((b * kJ + k2) * kN + n) * kC + h * kHC;
        float acc = 0.f;
#pragma unroll
        for (int c = 0; c < kHC; ++c) acc = fmaf(qrow[c], krow[c], acc);
        s[k2] = acc * kScale;
    }

    float mx = s[0];
#pragma unroll
    for (int k2 = 1; k2 < kJ; ++k2) mx = fmaxf(mx, s[k2]);
    float e[kJ], sum = 0.f;
#pragma unroll
    for (int k2 = 0; k2 < kJ; ++k2) { e[k2] = __expf(s[k2] - mx); sum += e[k2]; }
    float inv = 1.0f / sum;
    float* o1 = att1 + ((size_t)u * kJ + j) * kJ;
#pragma unroll
    for (int k2 = 0; k2 < kJ; ++k2) o1[k2] = e[k2] * inv;

#pragma unroll
    for (int k2 = 0; k2 < kJ; ++k2) {
        const float bias = 0.5f * ((As[j * kJ + k2] + adjs[j * kJ + k2]) +
                                   (As[k2 * kJ + j] + adjs[k2 * kJ + j]));
        s[k2] += bias;
    }
    mx = s[0];
#pragma unroll
    for (int k2 = 1; k2 < kJ; ++k2) mx = fmaxf(mx, s[k2]);
    sum = 0.f;
#pragma unroll
    for (int k2 = 0; k2 < kJ; ++k2) { e[k2] = __expf(s[k2] - mx); sum += e[k2]; }
    inv = 1.0f / sum;
    float* o2 = att2 + ((size_t)u * kJ + j) * kJ;
#pragma unroll
    for (int k2 = 0; k2 < kJ; ++k2) o2[k2] = e[k2] * inv;
}

// ---------------------------------------------------------------------------
// Kernel 3: variate attention. One wave per (b,h,j); lane n owns row n.
// ---------------------------------------------------------------------------
__global__ __launch_bounds__(256)
void att_v_kernel(const float* __restrict__ Q, const float* __restrict__ Kt,
                  const float* __restrict__ Av, const float* __restrict__ adjv,
                  float* __restrict__ att1, float* __restrict__ att2)
{
    const int lane = threadIdx.x & 31;
    const int u = blockIdx.x * 8 + (threadIdx.x >> 5);  // (b*H + h)*J + j
    const int j = u % kJ;
    const int h = (u / kJ) % kH;
    const int b = u / (kJ * kH);
    const int n = lane;
    if (n >= kN) return;

    const float* qrow = Q + (size_t)((b * kJ + j) * kN + n) * kC + h * kHC;
    float s[kN];
    for (int m = 0; m < kN; ++m) {
        const float* krow = Kt + (size_t)((b * kJ + j) * kN + m) * kC + h * kHC;
        float acc = 0.f;
#pragma unroll
        for (int c = 0; c < kHC; ++c) acc = fmaf(qrow[c], krow[c], acc);
        s[m] = acc * kScale;
    }

    float mx = s[0];
#pragma unroll
    for (int m = 1; m < kN; ++m) mx = fmaxf(mx, s[m]);
    float e[kN], sum = 0.f;
#pragma unroll
    for (int m = 0; m < kN; ++m) { e[m] = __expf(s[m] - mx); sum += e[m]; }
    float inv = 1.0f / sum;
    float* o1 = att1 + ((size_t)u * kN + n) * kN;
#pragma unroll
    for (int m = 0; m < kN; ++m) o1[m] = e[m] * inv;

#pragma unroll
    for (int m = 0; m < kN; ++m) {
        const float bias = 0.5f * ((Av[n * kN + m] + adjv[n * kN + m]) +
                                   (Av[m * kN + n] + adjv[m * kN + n]));
        s[m] += bias;
    }
    mx = s[0];
#pragma unroll
    for (int m = 1; m < kN; ++m) mx = fmaxf(mx, s[m]);
    sum = 0.f;
#pragma unroll
    for (int m = 0; m < kN; ++m) { e[m] = __expf(s[m] - mx); sum += e[m]; }
    inv = 1.0f / sum;
    float* o2 = att2 + ((size_t)u * kN + n) * kN;
#pragma unroll
    for (int m = 0; m < kN; ++m) o2[m] = e[m] * inv;
}

// ---------------------------------------------------------------------------
// Kernel 4: fused combine. One wave per (b,h,j,n); lane = channel c (0..31).
// ---------------------------------------------------------------------------
__global__ __launch_bounds__(256)
void combine_kernel(const float* __restrict__ att_s, const float* __restrict__ att_s2,
                    const float* __restrict__ att_v, const float* __restrict__ att_v2,
                    const float* __restrict__ vs, const float* __restrict__ vv,
                    const float* __restrict__ vsv, float* __restrict__ Y)
{
    const int lane = threadIdx.x & 31;
    const int u = blockIdx.x * 8 + (threadIdx.x >> 5);  // ((b*H+h)*J + j)*N + n
    const int n = u % kN;
    const int j = (u / kN) % kJ;
    const int h = (u / (kN * kJ)) % kH;
    const int b = u / (kN * kJ * kH);
    const int us = (b * kH + h) * kN + n;   // att_s row base (b,h,n)
    const int uv = (b * kH + h) * kJ + j;   // att_v row base (b,h,j)
    const int c = lane;

    float avrow[kN];
#pragma unroll
    for (int m = 0; m < kN; ++m)
        avrow[m] = att_v[((size_t)uv * kN + n) * kN + m];

    float acc_vs = 0.f, acc_vv = 0.f, acc_vsv = 0.f;
    for (int k2 = 0; k2 < kJ; ++k2) {
        const float wgt = att_s2[((size_t)us * kJ + j) * kJ + k2];
        acc_vs = fmaf(wgt, vs[(size_t)((b * kJ + k2) * kN + n) * kC + h * kHC + c], acc_vs);
    }
    for (int m = 0; m < kN; ++m) {
        const float wgt = att_v2[((size_t)uv * kN + n) * kN + m];
        acc_vv = fmaf(wgt, vv[(size_t)((b * kJ + j) * kN + m) * kC + h * kHC + c], acc_vv);
    }
    for (int k2 = 0; k2 < kJ; ++k2) {
        float inner = 0.f;
#pragma unroll
        for (int m = 0; m < kN; ++m)
            inner = fmaf(avrow[m],
                         vsv[(size_t)((b * kJ + k2) * kN + m) * kC + h * kHC + c],
                         inner);
        acc_vsv = fmaf(att_s[((size_t)us * kJ + j) * kJ + k2], inner, acc_vsv);
    }

    const size_t row = (size_t)(b * kJ + j) * kN + n;
    Y[row * kCat +          h * kHC + c] = acc_vsv;
    Y[row * kCat + kC     + h * kHC + c] = acc_vs;
    Y[row * kCat + 2 * kC + h * kHC + c] = acc_vv;
}

// ---------------------------------------------------------------------------
// Kernel 5: out = Y @ W_proj + b_proj  (bf16 WMMA, f32 accum), LDS-staged.
// Grid: 289 (M blocks) * 2 (N blocks) = 578 blocks; K = 768 -> 24 steps.
// ---------------------------------------------------------------------------
__global__ __launch_bounds__(256)
void proj_gemm_kernel(const float* __restrict__ Y,
                      const float* __restrict__ Wp,
                      const float* __restrict__ bp,
                      float* __restrict__ out)
{
    __shared__ __bf16 As[64][kLdsPad];
    __shared__ __bf16 Bs[128][kLdsPad];

    const int t    = threadIdx.x;
    const int lane = t & 31;
    const int w    = t >> 5;
    const int wm   = w & 3;
    const int wn   = w >> 2;
    const int mlo  = lane & 15;
    const bool hi  = lane >= 16;
    const int kgA  = hi ? 8 : 0;
    const int kgB  = hi ? 16 : 0;

    const int bm = blockIdx.x >> 1;
    const int bn = blockIdx.x & 1;
    const int rb = bm * 64;
    const int cb = bn * 128;

    const int arow = t >> 2;
    const int akc  = (t & 3) * 8;
    const int bcl  = t & 127;
    const int bkh  = (t >> 7) * 16;

    v8f acc[4] = {};

    for (int kt = 0; kt < kCat; kt += 32) {
        __syncthreads();
        {
            const float* src = Y + (size_t)(rb + arow) * kCat + kt + akc;
#pragma unroll
            for (int i = 0; i < 8; ++i)
                As[arow][akc + i] = (__bf16)src[i];
            const int ktn = (kt + 32 < kCat) ? kt + 32 : 0;
            __builtin_prefetch(Y + (size_t)(rb + arow) * kCat + ktn + akc, 0, 1);
        }
        {
#pragma unroll
            for (int i = 0; i < 16; ++i)
                Bs[bcl][bkh + i] =
                    (__bf16)Wp[(size_t)(kt + bkh + i) * kC + cb + bcl];
            const int ktn = (kt + 32 < kCat) ? kt + 32 : 0;
            __builtin_prefetch(Wp + (size_t)(ktn + bkh) * kC + cb + bcl, 0, 1);
        }
        __syncthreads();

        v16bf a;
#pragma unroll
        for (int e = 0; e < 16; ++e) {
            const int ka = (e < 8) ? (kgA + e) : (16 + kgA + (e - 8));
            a[e] = As[wm * 16 + mlo][ka];
        }
#pragma unroll
        for (int f = 0; f < 4; ++f) {
            v16bf b;
#pragma unroll
            for (int e = 0; e < 16; ++e)
                b[e] = Bs[wn * 64 + f * 16 + mlo][(e < 8) ? (kgA + e) : (16 + kgA + (e - 8))];
            acc[f] = __builtin_amdgcn_wmma_f32_16x16x32_bf16(
                false, a, false, b, (short)0, acc[f], false, false);
        }
    }

#pragma unroll
    for (int f = 0; f < 4; ++f) {
        const int col = cb + wn * 64 + f * 16 + mlo;
        const float bias = bp[col];
#pragma unroll
        for (int i = 0; i < 8; ++i) {
            const int row = rb + wm * 16 + i + (hi ? 8 : 0);
            out[(size_t)row * kC + col] = acc[f][i] + bias;
        }
    }
}

// ---------------------------------------------------------------------------
extern "C" void kernel_launch(void* const* d_in, const int* in_sizes, int n_in,
                              void* d_out, int out_size, void* d_ws, size_t ws_size,
                              hipStream_t stream)
{
    const float* X     = (const float*)d_in[0];
    const float* A_s   = (const float*)d_in[1];
    const float* A_v   = (const float*)d_in[2];
    const float* Wqkv  = (const float*)d_in[3];
    const float* Wproj = (const float*)d_in[4];
    const float* bproj = (const float*)d_in[5];
    const float* Mm    = (const float*)d_in[6];
    const float* adj_s = (const float*)d_in[7];
    const float* adj_v = (const float*)d_in[8];
    float* out = (float*)d_out;

    float* ws = (float*)d_ws;
    const size_t rc    = (size_t)kRows * kC;
    const size_t attsz = (size_t)kBT * kH * kJ * kJ * kN;
    float* q      = ws;
    float* kk     = ws + rc;
    float* vs     = ws + 2 * rc;
    float* vv     = ws + 3 * rc;
    float* vsv    = ws + 4 * rc;
    float* att_s  = ws + 5 * rc;
    float* att_s2 = att_s + attsz;
    float* att_v  = att_s2 + attsz;
    float* att_v2 = att_v + attsz;
    float* Y      = att_v2 + attsz;

    const int blocks1 = (kRows / 64) * (kQKV / 128);  // 289 * 10 = 2890
    const int waves2  = kBT * kH * kN;                // 8704
    const int waves3  = kBT * kH * kJ;                // 8704
    const int waves4  = kBT * kH * kJ * kN;           // 147968
    const int blocks5 = (kRows / 64) * (kC / 128);    // 289 * 2 = 578

    qkv_gemm_kernel<<<dim3(blocks1), dim3(256), 0, stream>>>(
        X, Wqkv, Mm, q, kk, vs, vv, vsv);
    att_s_kernel<<<dim3(waves2 / 8), dim3(256), 0, stream>>>(
        q, kk, A_s, adj_s, att_s, att_s2);
    att_v_kernel<<<dim3(waves3 / 8), dim3(256), 0, stream>>>(
        q, kk, A_v, adj_v, att_v, att_v2);
    combine_kernel<<<dim3(waves4 / 8), dim3(256), 0, stream>>>(
        att_s, att_s2, att_v, att_v2, vs, vv, vsv, Y);
    proj_gemm_kernel<<<dim3(blocks5), dim3(256), 0, stream>>>(
        Y, Wproj, bproj, out);
}
